// Encoder_49727131353307
// MI455X (gfx1250) — matrix operand order repaired and checked by reading, not compiled, not run on hardware
//
#include <hip/hip_runtime.h>
#include <hip/hip_bf16.h>

typedef __attribute__((ext_vector_type(16))) __bf16 v16bf;
typedef __attribute__((ext_vector_type(8)))  float  v8f;

namespace {

constexpr int B  = 64;
constexpr int T  = 512;
constexpr int D  = 256;
constexpr int H  = 1024;
constexpr int H3 = 3 * H;   // 3072

// ---- helpers ---------------------------------------------------------------

__device__ inline __bf16 f2bf(float f) {       // RNE-ish (cold paths)
  union { float f; unsigned u; } v; v.f = f;
  unsigned r = (v.u + 0x7FFFu + ((v.u >> 16) & 1u)) >> 16;
  unsigned short s = (unsigned short)r;
  return __builtin_bit_cast(__bf16, s);
}

// round-half-up pack of two fp32 into packed bf16x2 (2 VALU/elem)
__device__ inline unsigned pack_bf16x2(float a, float b) {
  union { float f; unsigned u; } x, y; x.f = a; y.f = b;
  return ((x.u + 0x8000u) >> 16) | ((y.u + 0x8000u) & 0xffff0000u);
}

struct U32B { uint4 lo, hi; };   // 32 bytes == one v16bf fragment

// A fragment from a row-major [16 x >=32] bf16 tile (rows 16B-aligned).
// ISA A-layout: lanes 0-15: M=lane, K in {0..7,16..23}; lanes 16-31: M=lane-16,
// K in {8..15,24..31}: two contiguous 16B runs at elem offsets {kb, 16+kb}.
__device__ inline v16bf load_a_frag(const __bf16* base, int strideElems, int lane) {
  int m   = lane & 15;
  int sel = lane >> 4;                       // kb = 8*sel
  const uint4* row = (const uint4*)(base + m * strideElems);
  U32B u;
  u.lo = row[sel];
  u.hi = row[2 + sel];
  return __builtin_bit_cast(v16bf, u);
}

// B fragment from a TRANSPOSED tile stored N-major: Bt[n][k], K-chunk of 32.
// Lane holds N=lane&15, K in {sel*8..sel*8+7, 16+sel*8..16+sel*8+7} (matches A).
__device__ inline v16bf load_bT_frag(const __bf16* baseT, int strideElems, int lane) {
  int n   = lane & 15;
  int sel = lane >> 4;
  const uint4* p = (const uint4*)(baseT + n * strideElems + sel * 8);
  U32B u;
  u.lo = p[0];
  u.hi = p[2];
  return __builtin_bit_cast(v16bf, u);
}

__device__ inline float sigmoidf_fast(float x) {
  return 1.0f / (1.0f + __expf(-x));
}

// ---- Kernel 0a: one-time fp32 -> bf16 of initial hidden state --------------

__global__ void __launch_bounds__(256)
hinit_kernel(const float* __restrict__ src, __bf16* __restrict__ dst) {
  int i = blockIdx.x * 256 + threadIdx.x;
  dst[i] = f2bf(src[i]);
}

// ---- Kernel 0b: embedding gather + pack: xb[B*T, D] = bf16(emb[tokens]) ----

__global__ void __launch_bounds__(256)
gather_x_kernel(const int* __restrict__ tokens, const float* __restrict__ emb,
                __bf16* __restrict__ xb) {
  int i = blockIdx.x * 256 + threadIdx.x;    // one thread per 8 elems
  int r = i >> 5, part = i & 31;             // 32 threads per row of 256
  const float4* src = (const float4*)(emb + (size_t)tokens[r] * D + part * 8);
  float4 v0 = src[0], v1 = src[1];
  uint4 o;
  o.x = pack_bf16x2(v0.x, v0.y);
  o.y = pack_bf16x2(v0.z, v0.w);
  o.z = pack_bf16x2(v1.x, v1.y);
  o.w = pack_bf16x2(v1.z, v1.w);
  *(uint4*)(xb + (size_t)r * D + part * 8) = o;
}

// ---- Kernel 0c: pack input-projection weights to bf16 ----------------------

__global__ void __launch_bounds__(256)
pack_wk_kernel(const float* __restrict__ wk, unsigned* __restrict__ wkbu) {
  int i = blockIdx.x * 256 + threadIdx.x;    // one thread per 2 elems
  float2 v = *(const float2*)(wk + (size_t)i * 2);
  wkbu[i] = pack_bf16x2(v.x, v.y);
}

// ---- Kernel 1: x_proj[B*T, 3H] = xb @ wkb + bias0 (bf16 WMMA, f32 acc) -----

constexpr int K1_BM = 128, K1_BN = 64, K1_BK = 32;
constexpr int ASTR  = 48;   // A row stride (elems): 96B rows, 16B aligned
constexpr int BTSTR = 40;   // transposed-B row stride (elems): 80B rows

__global__ void __launch_bounds__(256)
xproj_kernel(const __bf16* __restrict__ xb, const __bf16* __restrict__ wkb,
             const float* __restrict__ bias0, float* __restrict__ xproj) {
  __shared__ __bf16 As[2][K1_BM * ASTR];
  __shared__ __bf16 Bt[2][K1_BN * BTSTR];

  const int tid  = threadIdx.x;
  const int lane = tid & 31, wave = tid >> 5;
  const int nBase = blockIdx.x * K1_BN;
  const int mBase = blockIdx.y * K1_BM;

  v8f acc[4];
#pragma unroll
  for (int q = 0; q < 4; ++q)
#pragma unroll
    for (int i = 0; i < 8; ++i) acc[q][i] = 0.0f;

  auto stage = [&](int kc, int buf) {
    // A: pure aligned 16B copies from pre-gathered bf16 X (coalesced)
    {
      int r = tid >> 1, half = tid & 1;    // 128 rows x 2 halves of 16 elems
      const uint4* src =
          (const uint4*)(xb + (size_t)(mBase + r) * D + kc * K1_BK + half * 16);
      uint4* dst = (uint4*)(&As[buf][r * ASTR + half * 16]);
      dst[0] = src[0];
      dst[1] = src[1];
    }
    // B: bf16 pair reads (coalesced), transposed scatter into LDS; 4 fixed trips
#pragma unroll
    for (int it = 0; it < (K1_BK * K1_BN) / 2 / 256; ++it) {
      int i = tid + it * 256;
      int k = i >> 5, np = i & 31;
      unsigned u = *(const unsigned*)(wkb + (size_t)(kc * K1_BK + k) * H3 +
                                      nBase + np * 2);
      Bt[buf][(np * 2 + 0) * BTSTR + k] =
          __builtin_bit_cast(__bf16, (unsigned short)(u & 0xffffu));
      Bt[buf][(np * 2 + 1) * BTSTR + k] =
          __builtin_bit_cast(__bf16, (unsigned short)(u >> 16));
    }
  };

  stage(0, 0);
  __syncthreads();

  constexpr int NKC = D / K1_BK;   // 8
  for (int kc = 0; kc < NKC; ++kc) {
    int cur = kc & 1;
    if (kc + 1 < NKC) stage(kc + 1, cur ^ 1);

    v16bf a = load_a_frag(&As[cur][wave * 16 * ASTR], ASTR, lane);
#pragma unroll
    for (int nt = 0; nt < 4; ++nt) {
      v16bf b = load_bT_frag(&Bt[cur][nt * 16 * BTSTR], BTSTR, lane);
      acc[nt] = __builtin_amdgcn_wmma_f32_16x16x32_bf16(
          false, a, false, b, (short)0, acc[nt], false, false);
    }
    __syncthreads();
  }

  // C layout: VGPR i -> M=i (lanes 0-15) / M=i+8 (lanes 16-31); N = lane&15
  const int mRow  = mBase + wave * 16 + (lane >> 4) * 8;
  const int nCol0 = nBase + (lane & 15);
#pragma unroll
  for (int nt = 0; nt < 4; ++nt) {
    int col = nCol0 + nt * 16;
    float bz = bias0[col];
#pragma unroll
    for (int i = 0; i < 8; ++i)
      xproj[(size_t)(mRow + i) * H3 + col] = acc[nt][i] + bz;
  }
}

// ---- Kernel 2: persistent GRU scan -----------------------------------------
// 32 WGs; WG w owns h columns [32w, 32w+32). rec_kernel gate slices resident
// in LDS as transposed bf16 (198KB). h kept ping-ponged in BOTH fp32 (for the
// recurrence state) and bf16 (matmul operand; staged as pure 16B copies).

constexpr int NWG    = 32;
constexpr int CW     = H / NWG;      // 32 columns per WG
constexpr int NL     = 3 * CW;       // 96 local rec columns (z|r|h)
constexpr int WTSTR  = H + 8;        // 1032 elems: 2064B rows (129*16B)
constexpr int HSTR   = 48;           // h-chunk A rows: 96B
constexpr int RSTR   = NL + 4;       // 100
constexpr int HS_ELE = B * HSTR;     // one h-chunk buffer

__global__ void __launch_bounds__(256)
gru_scan_kernel(const float* __restrict__ xproj, const float* __restrict__ wr,
                const float* __restrict__ bias1,
                float* __restrict__ h0f, float* __restrict__ h1f,
                __bf16* __restrict__ h0b, __bf16* __restrict__ h1b,
                float* __restrict__ out, float* __restrict__ state,
                unsigned* __restrict__ ctr) {
  extern __shared__ char smem[];
  __bf16* Wt = (__bf16*)smem;                                   // NL * WTSTR
  __bf16* Hs = (__bf16*)(smem + (size_t)NL * WTSTR * 2);        // 2 * HS_ELE
  float*  Rs = (float*)(smem + (size_t)NL * WTSTR * 2 + 2 * HS_ELE * 2); // B*RSTR

  const int tid  = threadIdx.x;
  const int lane = tid & 31, wave = tid >> 5;
  const int c0   = blockIdx.x * CW;

  // One-time: this WG's three gate column slices of rec_kernel, transposed.
  for (int i = tid; i < H * NL; i += 256) {
    int k = i / NL, n = i % NL;
    int g = n >> 5, cc = n & 31;
    Wt[(size_t)n * WTSTR + k] = f2bf(wr[(size_t)k * H3 + g * H + c0 + cc]);
  }
  __syncthreads();

  // Tile map: 24 tiles (4 M x 6 N), 3 per wave sharing one M-tile.
  const int mt  = wave >> 1;           // 0..3
  const int ntb = (wave & 1) * 3;      // 0 or 3

  float*  hf[2] = { h0f, h1f };
  __bf16* hb[2] = { h0b, h1b };
  unsigned barr = 0;

  // h staging geometry: 1 aligned uint4 (8 bf16) per thread per chunk.
  const int sm = tid >> 2, spart = tid & 3;
  // gate-update geometry
  const int ub0 = tid >> 5, uj = tid & 31;

  for (int t = 0; t < T; ++t) {
    const float*  hinf  = hf[t & 1];
    float*        houtf = hf[(t + 1) & 1];
    const __bf16* hinb  = hb[t & 1];
    __bf16*       houtb = hb[(t + 1) & 1];

    v8f acc[3];
#pragma unroll
    for (int q = 0; q < 3; ++q)
#pragma unroll
      for (int i = 0; i < 8; ++i) acc[q][i] = 0.0f;

    auto stageH = [&](int kc, int buf) {
      const uint4* src = (const uint4*)(hinb + sm * H + kc * 32 + spart * 8);
      uint4* dst = (uint4*)(Hs + buf * HS_ELE + sm * HSTR + spart * 8);
      *dst = *src;
    };

    stageH(0, 0);

    // Prefetch this step's x_proj operands (384 64B lines per WG) so the
    // gate update ~3us later hits warm lines (global_prefetch_b8).
#pragma unroll
    for (int pp = 0; pp < 2; ++pp) {
      int p = tid + pp * 256;          // (b, gate, half) over 64 x 3 x 2
      int b = p >> 3, u = p & 7;
      if (u < 6) {
        int gate = u >> 1, half = u & 1;
        const float* addr =
            xproj + ((size_t)b * T + t) * (size_t)H3 + gate * H + c0 + half * 16;
        __builtin_prefetch(addr, 0, 1);
      }
    }
    __syncthreads();

    constexpr int NKC = H / 32;   // 32
    for (int kc = 0; kc < NKC; ++kc) {
      int cur = kc & 1;
      if (kc + 1 < NKC) stageH(kc + 1, cur ^ 1);

      v16bf a = load_a_frag(&Hs[cur * HS_ELE + mt * 16 * HSTR], HSTR, lane);
#pragma unroll
      for (int q = 0; q < 3; ++q) {
        int nt = ntb + q;
        v16bf b = load_bT_frag(&Wt[(size_t)(nt * 16) * WTSTR + kc * 32],
                               WTSTR, lane);
        acc[q] = __builtin_amdgcn_wmma_f32_16x16x32_bf16(
            false, a, false, b, (short)0, acc[q], false, false);
      }
      __syncthreads();
    }

    // spill rec tiles to LDS for cross-wave gate fusion
#pragma unroll
    for (int q = 0; q < 3; ++q) {
      int nt = ntb + q;
      int m = mt * 16 + (lane >> 4) * 8;
      int n = nt * 16 + (lane & 15);
#pragma unroll
      for (int i = 0; i < 8; ++i) Rs[(m + i) * RSTR + n] = acc[q][i];
    }
    __syncthreads();

    // fused gate update over this WG's 64 x 32 slice of h; 8 fixed trips
#pragma unroll
    for (int it = 0; it < (B * CW) / 256; ++it) {
      int b = ub0 + it * 8;
      int col = c0 + uj;
      size_t xrow = ((size_t)b * T + t) * (size_t)H3;
      float xz = xproj[xrow + col];
      float xr = xproj[xrow + H + col];
      float xh = xproj[xrow + 2 * H + col];
      float rz = Rs[b * RSTR + uj]          + bias1[col];
      float rr = Rs[b * RSTR + CW + uj]     + bias1[H + col];
      float rh = Rs[b * RSTR + 2 * CW + uj] + bias1[2 * H + col];
      float z  = sigmoidf_fast(xz + rz);
      float r  = sigmoidf_fast(xr + rr);
      float hh = tanhf(xh + r * rh);
      float hold = hinf[b * H + col];
      float hn = z * hold + (1.0f - z) * hh;
      houtf[b * H + col] = hn;
      houtb[b * H + col] = f2bf(hn);
      out[((size_t)b * T + t) * (size_t)H + col] = hn;
      if (t == T - 1) state[b * H + col] = hn;
    }

    // grid-wide barrier: all WGs' hout visible before next step reads it
    barr += NWG;
    __syncthreads();
    if (tid == 0) {
      __threadfence();
      atomicAdd(ctr, 1u);
      while (__hip_atomic_load(ctr, __ATOMIC_RELAXED, __HIP_MEMORY_SCOPE_AGENT) < barr) {
        __builtin_amdgcn_s_sleep(1);
      }
      __threadfence();
    }
    __syncthreads();
  }
}

} // namespace

// ---- host ------------------------------------------------------------------

extern "C" void kernel_launch(void* const* d_in, const int* in_sizes, int n_in,
                              void* d_out, int out_size, void* d_ws, size_t ws_size,
                              hipStream_t stream) {
  (void)in_sizes; (void)n_in; (void)out_size; (void)ws_size;
  const int*   tokens = (const int*)d_in[0];
  const float* h_init = (const float*)d_in[1];
  const float* emb    = (const float*)d_in[2];
  const float* wk     = (const float*)d_in[3];
  const float* wr     = (const float*)d_in[4];
  const float* bias   = (const float*)d_in[5];

  float* out   = (float*)d_out;
  float* state = out + (size_t)B * T * H;

  char*   ws    = (char*)d_ws;
  float*  xproj = (float*)ws;
  size_t  off   = (size_t)B * T * H3 * sizeof(float);
  float*  h0f = (float*)(ws + off);  off += (size_t)B * H * sizeof(float);
  float*  h1f = (float*)(ws + off);  off += (size_t)B * H * sizeof(float);
  __bf16* h0b = (__bf16*)(ws + off); off += (size_t)B * H * sizeof(__bf16);
  __bf16* h1b = (__bf16*)(ws + off); off += (size_t)B * H * sizeof(__bf16);
  __bf16* xb  = (__bf16*)(ws + off); off += (size_t)B * T * D * sizeof(__bf16);
  __bf16* wkb = (__bf16*)(ws + off); off += (size_t)D * H3 * sizeof(__bf16);
  unsigned* ctr = (unsigned*)(ws + off);

  hipMemsetAsync(ctr, 0, 64, stream);
  hipMemcpyAsync(h0f, h_init, (size_t)B * H * sizeof(float),
                 hipMemcpyDeviceToDevice, stream);
  hinit_kernel<<<(B * H) / 256, 256, 0, stream>>>(h_init, h0b);

  gather_x_kernel<<<(B * T * D / 8) / 256, 256, 0, stream>>>(tokens, emb, xb);
  pack_wk_kernel<<<(D * H3 / 2) / 256, 256, 0, stream>>>(wk, (unsigned*)wkb);

  dim3 grid1(H3 / K1_BN, (B * T) / K1_BM);   // 48 x 256
  xproj_kernel<<<grid1, 256, 0, stream>>>(xb, wkb, bias, xproj);

  size_t smem = (size_t)NL * WTSTR * 2 + (size_t)2 * HS_ELE * 2 + (size_t)B * RSTR * 4;
  gru_scan_kernel<<<NWG, 256, smem, stream>>>(xproj, wr, bias + H3,
                                              h0f, h1f, h0b, h1b,
                                              out, state, ctr);
}
